// VanillaRNN_19344532701614
// MI455X (gfx1250) — matrix operand order repaired and checked by reading, not compiled, not run on hardware
//
#include <hip/hip_runtime.h>

#define TT 1024
#define BB 256
#define DD 256
#define HH 512
#define OO 256
#define HP 520   // padded LDS row (multiple of 8 -> 16B-aligned lane loads)

typedef __attribute__((ext_vector_type(16))) __bf16 bf16x16;
typedef __attribute__((ext_vector_type(8)))  float  f32x8;

union BFV { bf16x16 v; uint4 q[2]; };
union F8V { f32x8 v; float4 q[2]; };

__device__ __forceinline__ unsigned short f2bf(float f) {
    unsigned int u = __float_as_uint(f);
    unsigned int r = u + 0x7FFFu + ((u >> 16) & 1u);   // round-to-nearest-even
    return (unsigned short)(r >> 16);
}

// B operand: chunks at +0 and +16 elements (two ds_load_b128)
#define LOADB(dst, P, K) do { const uint4* p_ = (const uint4*)((P) + (K)); \
                              (dst).q[0] = p_[0]; (dst).q[1] = p_[2]; } while (0)
// A operand: chunks at +0 and +16 elements (two global_load_b128)
#define LOADA(dst, P, K) do { const uint4* p_ = (const uint4*)((P) + (K)); \
                              (dst).q[0] = p_[0]; (dst).q[1] = p_[2]; } while (0)
#define WMMA_BF16(a, b, c) __builtin_amdgcn_wmma_f32_16x16x32_bf16( \
                              false, (a).v, false, (b).v, (short)0, (c), false, false)

// ---- one-shot f32 -> bf16 conversion of W and V into workspace ----
__global__ __launch_bounds__(512)
void rnn_prep_bf16(const float* __restrict__ W, const float* __restrict__ V,
                   unsigned short* __restrict__ Wb, unsigned short* __restrict__ Vb) {
    int i = blockIdx.x * blockDim.x + threadIdx.x;
    if (i < HH * HH) Wb[i] = f2bf(W[i]);
    if (i < OO * HH) Vb[i] = f2bf(V[i]);
}

// ---- persistent RNN kernel: 1 block = 16 batch columns, runs all T steps ----
__global__ __launch_bounds__(512)
void rnn_persistent(const int* __restrict__ tokens,
                    const float* __restrict__ h0,
                    const float* __restrict__ U,
                    const float* __restrict__ bh,
                    const float* __restrict__ by,
                    const unsigned short* __restrict__ Wb,
                    const unsigned short* __restrict__ Vb,
                    float* __restrict__ out) {
    // double-buffered h (transposed, bf16): h_t lives in htB[t&1]
    __shared__ __align__(16) unsigned short htB[2][16][HP];
    __shared__ float bhS[HH];
    __shared__ float byS[OO];
    __shared__ int   tokS[2][16];

    const int tid  = threadIdx.x;
    const int lane = tid & 31;           // wave32
    const int wv   = tid >> 5;           // wave id 0..15
    const int ln   = lane & 15;          // N (batch col) within tile
    const int hi   = lane >> 4;          // half-wave select
    const int bcol = blockIdx.x * 16;    // this WG's batch columns

    bhS[tid] = bh[tid];
    if (tid < OO) byS[tid] = by[tid];
    if (tid < 16) tokS[0][tid] = tokens[bcol + tid];     // x_0
    // h_{-1} = h0 goes into buffer parity (0-1)&1 == 1
    for (int i = tid; i < 16 * HH; i += 512) {
        int n = i >> 9, k = i & (HH - 1);
        htB[1][n][k] = f2bf(h0[(size_t)k * BB + bcol + n]);
    }

    const int m0 = wv * 32;  // this wave's two W@h output tiles: rows m0..m0+31
    const unsigned short* Aw0 = Wb + (size_t)(m0 + ln) * HH + hi * 8;
    const unsigned short* Aw1 = Aw0 + (size_t)16 * HH;
    const unsigned short* Av  = Vb + (size_t)(wv * 16 + ln) * HH + hi * 8;
    const unsigned short* Bb0 = &htB[0][ln][hi * 8];
    const unsigned short* Bb1 = &htB[1][ln][hi * 8];

    float* outT = out + (size_t)TT * BB * OO;  // hT section of d_out

    __syncthreads();   // htB[1] = h0, tokS[0] valid

#pragma unroll 1
    for (int t = 0; t < TT; ++t) {
        const unsigned short* Brd = (t & 1) ? Bb0 : Bb1;   // h_{t-1}
        const unsigned short* Bwr = (t & 1) ? Bb1 : Bb0;   // will hold h_t (via htB write)
        const int wr = t & 1;

        // prefetch next step's tokens (visible after this iteration's barrier)
        if (tid < 16 && t + 1 < TT)
            tokS[(t + 1) & 1][tid] = tokens[(size_t)(t + 1) * BB + bcol + tid];
        (void)Bwr;

        // issue U gathers early so they retire under the WMMA chain
        const int tk = tokS[t & 1][ln];
        float ux0[8], ux1[8];
#pragma unroll
        for (int v = 0; v < 8; ++v) {
            ux0[v] = U[(size_t)(m0 + hi * 8 + v) * DD + tk];
            ux1[v] = U[(size_t)(m0 + 16 + hi * 8 + v) * DD + tk];
        }

        // ---- phase A: Wh = W @ h_{t-1}, two 16x16 tiles per wave, pipelined ----
        f32x8 acc0 = {}; f32x8 acc1 = {};
        {
            BFV bc, bn, a0c, a0n, a1c, a1n;
            LOADB(bc, Brd, 0);
            LOADA(a0c, Aw0, 0);
            LOADA(a1c, Aw1, 0);
#pragma unroll
            for (int k0 = 0; k0 < HH; k0 += 32) {
                const int kn = k0 + 32;
                if (kn < HH) {                 // issue next chunk before consuming current
                    LOADB(bn, Brd, kn);
                    LOADA(a0n, Aw0, kn);
                    LOADA(a1n, Aw1, kn);
                }
                acc0 = WMMA_BF16(a0c, bc, acc0);
                acc1 = WMMA_BF16(a1c, bc, acc1);
                if (kn < HH) { bc = bn; a0c = a0n; a1c = a1n; }
            }
        }

        // ---- h_t = tanh(Wh + U[:, x_t] + bh) ----
        float hv0[8], hv1[8];
#pragma unroll
        for (int v = 0; v < 8; ++v) {
            int ma = m0 + hi * 8 + v;
            int mb = ma + 16;
            hv0[v] = tanhf(acc0[v] + ux0[v] + bhS[ma]);
            hv1[v] = tanhf(acc1[v] + ux1[v] + bhS[mb]);
        }
        if (t == TT - 1) {  // emit hT (f32) once
#pragma unroll
            for (int v = 0; v < 8; ++v) {
                outT[(size_t)(m0 + hi * 8 + v)      * BB + bcol + ln] = hv0[v];
                outT[(size_t)(m0 + 16 + hi * 8 + v) * BB + bcol + ln] = hv1[v];
            }
        }

        // ---- write h_t (bf16, transposed) into the OTHER buffer: no read conflict ----
        uint4 p0, p1;
        p0.x = f2bf(hv0[0]) | ((unsigned)f2bf(hv0[1]) << 16);
        p0.y = f2bf(hv0[2]) | ((unsigned)f2bf(hv0[3]) << 16);
        p0.z = f2bf(hv0[4]) | ((unsigned)f2bf(hv0[5]) << 16);
        p0.w = f2bf(hv0[6]) | ((unsigned)f2bf(hv0[7]) << 16);
        p1.x = f2bf(hv1[0]) | ((unsigned)f2bf(hv1[1]) << 16);
        p1.y = f2bf(hv1[2]) | ((unsigned)f2bf(hv1[3]) << 16);
        p1.z = f2bf(hv1[4]) | ((unsigned)f2bf(hv1[5]) << 16);
        p1.w = f2bf(hv1[6]) | ((unsigned)f2bf(hv1[7]) << 16);
        *(uint4*)&htB[wr][ln][m0 + hi * 8]      = p0;      // ds_store_b128
        *(uint4*)&htB[wr][ln][m0 + 16 + hi * 8] = p1;

        __syncthreads();   // single barrier per step: htB[t&1] == h_t everywhere

        // ---- phase B: y = V @ h_t + by (V operands reloaded from global each step), pipelined ----
        const unsigned short* Brd2 = (t & 1) ? Bb1 : Bb0;  // h_t
        f32x8 yac = {};
        {
            BFV bc, bn, ac, an;
            LOADB(bc, Brd2, 0);
            LOADA(ac, Av, 0);
#pragma unroll
            for (int k0 = 0; k0 < HH; k0 += 32) {
                const int kn = k0 + 32;
                if (kn < HH) {
                    LOADB(bn, Brd2, kn);
                    LOADA(an, Av, kn);
                }
                yac = WMMA_BF16(ac, bc, yac);
                if (kn < HH) { bc = bn; ac = an; }
            }
        }
        const int o0 = wv * 16 + hi * 8;
        F8V y; y.v = yac;
#pragma unroll
        for (int v = 0; v < 8; ++v) y.v[v] += byS[o0 + v];
        float* op = out + (size_t)t * BB * OO + (size_t)(bcol + ln) * OO + o0;
        *(float4*)(op + 0) = y.q[0];                       // global_store_b128 x2
        *(float4*)(op + 4) = y.q[1];
    }
}

extern "C" void kernel_launch(void* const* d_in, const int* in_sizes, int n_in,
                              void* d_out, int out_size, void* d_ws, size_t ws_size,
                              hipStream_t stream) {
    (void)in_sizes; (void)n_in; (void)out_size; (void)ws_size;
    const int*   tokens = (const int*)  d_in[0];   // (T,B)
    const float* h0     = (const float*)d_in[1];   // (H,B)
    const float* U      = (const float*)d_in[2];   // (H,D)
    const float* W      = (const float*)d_in[3];   // (H,H)
    const float* V      = (const float*)d_in[4];   // (O,H)
    const float* bh     = (const float*)d_in[5];   // (H,1)
    const float* by     = (const float*)d_in[6];   // (O,1)
    float* out = (float*)d_out;

    unsigned short* Wb = (unsigned short*)d_ws;            // 512 KB
    unsigned short* Vb = Wb + (size_t)HH * HH;             // 256 KB

    rnn_prep_bf16<<<(HH * HH + 511) / 512, 512, 0, stream>>>(W, V, Wb, Vb);
    rnn_persistent<<<BB / 16, 512, 0, stream>>>(tokens, h0, U, bh, by, Wb, Vb, out);
}
